// Fusion_70257075028106
// MI455X (gfx1250) — matrix operand order repaired and checked by reading, not compiled, not run on hardware
//
#include <hip/hip_runtime.h>

typedef unsigned short u16;
typedef __attribute__((ext_vector_type(16))) __bf16 v16bf;
typedef __attribute__((ext_vector_type(8)))  float  v8f;

#define KDIM 512
#define E_N 6

__device__ __forceinline__ u16 f2bf(float f) {
    unsigned u = __float_as_uint(f);
    unsigned r = u + 0x7FFFu + ((u >> 16) & 1u);
    return (u16)(r >> 16);
}
__device__ __forceinline__ float lrelu(float t) { return t > 0.f ? t : 0.2f * t; }

__device__ __forceinline__ void atomicMaxF(float* addr, float val) {
    unsigned* ua = (unsigned*)addr;
    unsigned old = *ua;
    while (__uint_as_float(old) < val) {
        unsigned assumed = old;
        old = atomicCAS(ua, assumed, __float_as_uint(val));
        if (old == assumed) break;
    }
}

// ---------------- prep: waS[g][i] = sum_j W[g][i][j]*a_src[g][j], same for a_dst
__global__ __launch_bounds__(256) void k_wa(const float* __restrict__ W,
                                            const float* __restrict__ aS,
                                            const float* __restrict__ aD,
                                            float* __restrict__ waS, float* __restrict__ waD) {
    __shared__ float rs[256], rd[256];
    int b = blockIdx.x, tid = threadIdx.x;
    int g = b >> 9, i = b & 511;
    const float* wr = W + (size_t)(g * 512 + i) * 512;
    const float* as = aS + g * 512;
    const float* ad = aD + g * 512;
    float s = 0.f, d = 0.f;
    for (int j = tid; j < 512; j += 256) { float wv = wr[j]; s += wv * as[j]; d += wv * ad[j]; }
    rs[tid] = s; rd[tid] = d; __syncthreads();
    for (int off = 128; off; off >>= 1) {
        if (tid < off) { rs[tid] += rs[tid + off]; rd[tid] += rd[tid + off]; }
        __syncthreads();
    }
    if (!tid) { waS[g * 512 + i] = rs[0]; waD[g * 512 + i] = rd[0]; }
}

// ---------------- convert+transpose W (4x512x512 f32) -> Wt bf16 with Wt[g][n][k]=W[g][k][n]
__global__ void k_cvt(const float* __restrict__ W, u16* __restrict__ Wt) {
    int i = blockIdx.x * 256 + threadIdx.x;           // 4*512*512 total
    int g = i >> 18, rem = i & 262143, k = rem >> 9, n = rem & 511;
    Wt[g * 262144 + n * 512 + k] = f2bf(W[i]);
}

// ---------------- as_k1[i]=kn[i]·waS1 (blocks 0..511); exercise alphas asE/adE (blocks 512..535)
__global__ __launch_bounds__(256) void k_small(const float* __restrict__ kn,
                                               const float* __restrict__ exer,
                                               const float* __restrict__ waS,
                                               const float* __restrict__ waD,
                                               float* __restrict__ as_k1,
                                               float* __restrict__ asE, float* __restrict__ adE) {
    __shared__ float rs[256], rd[256];
    int b = blockIdx.x, tid = threadIdx.x;
    if (b < 512) {
        const float* xr = kn + (size_t)b * 512;
        const float* w1 = waS + 512;
        float s = 0.f;
        for (int j = tid; j < 512; j += 256) s += xr[j] * w1[j];
        rs[tid] = s; __syncthreads();
        for (int off = 128; off; off >>= 1) { if (tid < off) rs[tid] += rs[tid + off]; __syncthreads(); }
        if (!tid) as_k1[b] = rs[0];
    } else {
        int t = b - 512, g = t / 6, j = t % 6;
        const float* xr = exer + j * 512;
        const float* wsp = waS + g * 512;
        const float* wdp = waD + g * 512;
        float s = 0.f, d = 0.f;
        for (int jj = tid; jj < 512; jj += 256) { float xv = xr[jj]; s += xv * wsp[jj]; d += xv * wdp[jj]; }
        rs[tid] = s; rd[tid] = d; __syncthreads();
        for (int off = 128; off; off >>= 1) {
            if (tid < off) { rs[tid] += rs[tid + off]; rd[tid] += rd[tid + off]; }
            __syncthreads();
        }
        if (!tid) { asE[g * 6 + j] = rs[0]; adE[g * 6 + j] = rd[0]; }
    }
}

// ---------------- GAT2 (e_from_k): per-exercise segment softmax over 2048 edges -> zB bf16 [6][512]
__global__ __launch_bounds__(256) void k_zB(const float* __restrict__ kn,
                                            const float* __restrict__ exer,
                                            const int* __restrict__ srcKE,
                                            const float* __restrict__ as_k1,
                                            const float* __restrict__ asE1,
                                            const float* __restrict__ adE1,
                                            u16* __restrict__ zB) {
    int j = blockIdx.x, tid = threadIdx.x;
    __shared__ float red[256];
    __shared__ float zsh[512];
    __shared__ float sc[2];
    float aj = adE1[j];
    float eself = lrelu(asE1[j] + aj);
    float mx = -3.4e38f;
    for (int m = tid; m < 2048; m += 256)
        if (srcKE[m] == j) mx = fmaxf(mx, lrelu(as_k1[m >> 2] + aj));
    if (tid == 0) mx = fmaxf(mx, eself);
    red[tid] = mx; __syncthreads();
    for (int off = 128; off; off >>= 1) { if (tid < off) red[tid] = fmaxf(red[tid], red[tid + off]); __syncthreads(); }
    if (!tid) sc[0] = red[0];
    __syncthreads();
    mx = sc[0];
    float den = 0.f;
    for (int m = tid; m < 2048; m += 256)
        if (srcKE[m] == j) den += __expf(lrelu(as_k1[m >> 2] + aj) - mx);
    if (tid == 0) den += __expf(eself - mx);
    red[tid] = den; __syncthreads();
    for (int off = 128; off; off >>= 1) { if (tid < off) red[tid] += red[tid + off]; __syncthreads(); }
    if (!tid) sc[1] = red[0];
    __syncthreads();
    float inv = 1.f / sc[1];
    float aself = __expf(eself - mx) * inv;
    for (int c = tid; c < 512; c += 256) zsh[c] = aself * exer[j * 512 + c];
    __syncthreads();
    for (int m = 0; m < 2048; ++m) {
        if (srcKE[m] != j) continue;
        float wgt = __expf(lrelu(as_k1[m >> 2] + aj) - mx) * inv;
        const float* kr = kn + (size_t)(m >> 2) * 512;
        for (int c = tid; c < 512; c += 256) zsh[c] += wgt * kr[c];
    }
    __syncthreads();
    for (int c = tid; c < 512; c += 256) zB[j * 512 + c] = f2bf(zsh[c]);
}

// ---------------- GAT4 helpers
__global__ void k_initc(float* __restrict__ emax, float* __restrict__ den, float* __restrict__ num) {
    int i = blockIdx.x * 256 + threadIdx.x;
    if (i < 6) { emax[i] = -3.4e38f; den[i] = 0.f; }
    if (i < 3072) num[i] = 0.f;
}

__global__ __launch_bounds__(256) void k_asstu(const float* __restrict__ stu,
                                               const float* __restrict__ waS3,
                                               float* __restrict__ asStu) {
    int w = threadIdx.x >> 5, lane = threadIdx.x & 31;
    int row = blockIdx.x * 8 + w;
    const float* xr = stu + (size_t)row * 512;
    float s = 0.f;
    for (int j = lane; j < 512; j += 32) s += xr[j] * waS3[j];
    for (int off = 16; off; off >>= 1) s += __shfl_down(s, off);
    if (!lane) asStu[row] = s;
}

__global__ __launch_bounds__(256) void k_cmax(const int* __restrict__ srcUE,
                                              const float* __restrict__ asStu,
                                              const float* __restrict__ asE3,
                                              const float* __restrict__ adE3,
                                              float* __restrict__ emax) {
    int j = blockIdx.x % 6, chunk = blockIdx.x / 6;   // grid 96
    int start = chunk * 9375, end = min(start + 9375, 150000);
    float aj = adE3[j];
    float mx = -3.4e38f;
    for (int m = start + threadIdx.x; m < end; m += 256)
        if (srcUE[m] == j) mx = fmaxf(mx, lrelu(asStu[m / 3] + aj));
    if (chunk == 0 && threadIdx.x == 0) mx = fmaxf(mx, lrelu(asE3[j] + aj));
    __shared__ float red[256];
    red[threadIdx.x] = mx; __syncthreads();
    for (int off = 128; off; off >>= 1) {
        if (threadIdx.x < off) red[threadIdx.x] = fmaxf(red[threadIdx.x], red[threadIdx.x + off]);
        __syncthreads();
    }
    if (!threadIdx.x) atomicMaxF(emax + j, red[0]);
}

__global__ __launch_bounds__(256) void k_cacc(const int* __restrict__ srcUE,
                                              const float* __restrict__ stu,
                                              const float* __restrict__ asStu,
                                              const float* __restrict__ adE3,
                                              const float* __restrict__ emax,
                                              float* __restrict__ den,
                                              float* __restrict__ num) {
    int j = blockIdx.x % 6, chunk = blockIdx.x / 6;   // grid 384
    int start = chunk * 2344, end = min(start + 2344, 150000);
    int tid = threadIdx.x;
    __shared__ float znum[512];
    for (int c = tid; c < 512; c += 256) znum[c] = 0.f;
    __syncthreads();
    float aj = adE3[j], mx = emax[j], dloc = 0.f;
    for (int m = start; m < end; ++m) {
        if (srcUE[m] != j) continue;
        float wgt = __expf(lrelu(asStu[m / 3] + aj) - mx);
        dloc += wgt;
        const float* xr = stu + (size_t)(m / 3) * 512;
        for (int c = tid; c < 512; c += 256) znum[c] += wgt * xr[c];
    }
    __syncthreads();
    for (int c = tid; c < 512; c += 256) atomicAdd(num + j * 512 + c, znum[c]);
    if (!tid) atomicAdd(den + j, dloc);
}

__global__ __launch_bounds__(256) void k_cfix(const float* __restrict__ exer,
                                              const float* __restrict__ asE3,
                                              const float* __restrict__ adE3,
                                              const float* __restrict__ emax,
                                              const float* __restrict__ den,
                                              const float* __restrict__ num,
                                              u16* __restrict__ zC) {
    int j = blockIdx.x, tid = threadIdx.x;
    float wself = __expf(lrelu(asE3[j] + adE3[j]) - emax[j]);
    float inv = 1.f / (den[j] + wself);
    for (int c = tid; c < 512; c += 256)
        zC[j * 512 + c] = f2bf((num[j * 512 + c] + wself * exer[j * 512 + c]) * inv);
}

// ---------------- tiled bf16 WMMA GEMM: out[M x 512] = (z) @ Wt^T + bias (+ base)
// fused=1: build z rows on the fly from node features X, per-row GAT softmax (deg edges + self)
// fused=0: read A (bf16) from Abf
__global__ __launch_bounds__(256) void gemm_tile(
    const float* __restrict__ X, const u16* __restrict__ Abf, const u16* __restrict__ Wt,
    const float* __restrict__ bias, const float* __restrict__ base, const float* __restrict__ exer,
    const int* __restrict__ srcIdx, const float* __restrict__ asE,
    const float* __restrict__ waS, const float* __restrict__ waD,
    float* __restrict__ out, int M, int deg, int fused) {
    __shared__ __align__(16) u16 At[128][40];
    __shared__ __align__(16) u16 Bt[128][40];
    __shared__ float sEx[E_N * 512];
    __shared__ float sAl[128][5];
    __shared__ int   sSrc[128][4];

    int tid = threadIdx.x, lane = tid & 31, w = tid >> 5;
    int row0 = blockIdx.x * 128, n0 = blockIdx.y * 128;

    for (int i = tid; i < E_N * 512; i += 256) sEx[i] = exer[i];

    if (fused) {
        for (int i = 0; i < 16; ++i) {
            int mr = w * 16 + i;
            int row = row0 + mr;
            float s = 0.f, d = 0.f;
            if (row < M) {
                const float* xr = X + (size_t)row * KDIM;
                for (int j = lane; j < KDIM; j += 32) { float xv = xr[j]; s += xv * waS[j]; d += xv * waD[j]; }
            }
            for (int off = 16; off; off >>= 1) { s += __shfl_down(s, off); d += __shfl_down(d, off); }
            if (lane == 0 && row < M) {
                int   sr[4];
                float ev[5];
                for (int e = 0; e < deg; ++e) { sr[e] = srcIdx[row * deg + e]; ev[e] = lrelu(asE[sr[e]] + d); }
                ev[deg] = lrelu(s + d);
                float mxv = ev[deg];
                for (int e = 0; e < deg; ++e) mxv = fmaxf(mxv, ev[e]);
                float den = 0.f;
                for (int e = 0; e <= deg; ++e) { ev[e] = __expf(ev[e] - mxv); den += ev[e]; }
                float inv = 1.f / den;
                for (int e = 0; e < deg; ++e) { sAl[mr][e] = ev[e] * inv; sSrc[mr][e] = sr[e]; }
                sAl[mr][4] = ev[deg] * inv;
            }
        }
    }
    __syncthreads();

    v8f acc[4][2];
#pragma unroll
    for (int a = 0; a < 4; ++a)
#pragma unroll
        for (int b = 0; b < 2; ++b)
#pragma unroll
            for (int e = 0; e < 8; ++e) acc[a][b][e] = 0.f;

    int waveM = w & 1, waveN = w >> 1;
    int lg = lane >> 4, lm = lane & 15;

    for (int kc = 0; kc < KDIM; kc += 32) {
        // ---- stage A tile (128 x 32 bf16)
        {
            int m = tid >> 1, kb = (tid & 1) * 16;
            int row = row0 + m;
            if (fused) {
                float vals[16];
                if (row < M) {
                    const float* xr = X + (size_t)row * KDIM + kc + kb;
                    if (kc + 32 < KDIM) __builtin_prefetch((const void*)(xr + 32), 0, 1);
                    float aself = sAl[m][4];
#pragma unroll
                    for (int q = 0; q < 16; ++q) vals[q] = aself * xr[q];
                    for (int e = 0; e < deg; ++e) {
                        float a = sAl[m][e];
                        const float* er = &sEx[sSrc[m][e] * KDIM + kc + kb];
#pragma unroll
                        for (int q = 0; q < 16; ++q) vals[q] += a * er[q];
                    }
                } else {
#pragma unroll
                    for (int q = 0; q < 16; ++q) vals[q] = 0.f;
                }
#pragma unroll
                for (int q = 0; q < 16; ++q) At[m][kb + q] = f2bf(vals[q]);
            } else {
                if (row < M) {
                    const u16* ar = Abf + (size_t)row * KDIM + kc + kb;
                    if (kc + 32 < KDIM) __builtin_prefetch((const void*)(ar + 32), 0, 1);
#pragma unroll
                    for (int q = 0; q < 16; ++q) At[m][kb + q] = ar[q];
                } else {
#pragma unroll
                    for (int q = 0; q < 16; ++q) At[m][kb + q] = 0;
                }
            }
        }
        // ---- stage B tile (Wt rows n0..n0+127, K chunk)
        {
            int n = tid >> 1, kb = (tid & 1) * 16;
            const u16* wr = Wt + (size_t)(n0 + n) * KDIM + kc + kb;
            if (kc + 32 < KDIM) __builtin_prefetch((const void*)(wr + 32), 0, 1);
#pragma unroll
            for (int q = 0; q < 16; ++q) Bt[n][kb + q] = wr[q];
        }
        __syncthreads();

        v16bf bfrag[2];
#pragma unroll
        for (int ns = 0; ns < 2; ++ns) {
            const u16* p = &Bt[waveN * 32 + ns * 16 + lm][lg * 16];
            union { uint4 q[2]; v16bf v; } u;
            u.q[0] = *(const uint4*)p;
            u.q[1] = *(const uint4*)(p + 8);
            bfrag[ns] = u.v;
        }
#pragma unroll
        for (int ms = 0; ms < 4; ++ms) {
            const u16* p = &At[waveM * 64 + ms * 16 + lm][0];
            union { uint4 q[2]; v16bf v; } ua;
            ua.q[0] = *(const uint4*)(p + lg * 16);
            ua.q[1] = *(const uint4*)(p + lg * 16 + 8);
            v16bf afrag = ua.v;
#pragma unroll
            for (int ns = 0; ns < 2; ++ns) {
                acc[ms][ns] = __builtin_amdgcn_wmma_f32_16x16x32_bf16(
                    false, afrag, false, bfrag[ns], (short)0, acc[ms][ns], false, false);
            }
        }
        __syncthreads();
    }

    // ---- epilogue: C layout VGPR r -> (M = r + 8*lg, N = lm)
#pragma unroll
    for (int ms = 0; ms < 4; ++ms)
#pragma unroll
        for (int ns = 0; ns < 2; ++ns) {
            int gcol = n0 + waveN * 32 + ns * 16 + lm;
            float bv = bias[gcol];
#pragma unroll
            for (int r = 0; r < 8; ++r) {
                int grow = row0 + waveM * 64 + ms * 16 + lg * 8 + r;
                if (grow < M) {
                    float v = acc[ms][ns][r] + bv;
                    if (base) v += base[(size_t)grow * KDIM + gcol];
                    out[(size_t)grow * KDIM + gcol] = v;
                }
            }
        }
}

// ---------------- final exercise combine (softmax over {s1,s2})
__global__ __launch_bounds__(256) void k_exer(const float* __restrict__ exer,
                                              const float* __restrict__ B,
                                              const float* __restrict__ C,
                                              const float* __restrict__ aw,
                                              const float* __restrict__ ab,
                                              float* __restrict__ outE) {
    int j = blockIdx.x, tid = threadIdx.x;
    __shared__ float r1[256], r2[256], wsh[2];
    const float* aw1 = aw + 1024;
    const float* aw2 = aw + 2048;
    float s1 = 0.f, s2 = 0.f;
    for (int c = tid; c < 512; c += 256) {
        float ev = exer[j * 512 + c], bv = B[j * 512 + c], cv = C[j * 512 + c];
        s1 += ev * aw1[c] + bv * aw1[512 + c];
        s2 += ev * aw2[c] + cv * aw2[512 + c];
    }
    r1[tid] = s1; r2[tid] = s2; __syncthreads();
    for (int off = 128; off; off >>= 1) {
        if (tid < off) { r1[tid] += r1[tid + off]; r2[tid] += r2[tid + off]; }
        __syncthreads();
    }
    if (!tid) {
        float t1 = r1[0] + ab[1], t2 = r2[0] + ab[2];
        float mxx = fmaxf(t1, t2);
        float e1 = __expf(t1 - mxx), e2 = __expf(t2 - mxx);
        float inv = 1.f / (e1 + e2);
        wsh[0] = e1 * inv; wsh[1] = e2 * inv;
    }
    __syncthreads();
    float w1 = wsh[0], w2 = wsh[1];
    for (int c = tid; c < 512; c += 256)
        outE[j * 512 + c] = exer[j * 512 + c] + w1 * B[j * 512 + c] + w2 * C[j * 512 + c];
}

extern "C" void kernel_launch(void* const* d_in, const int* in_sizes, int n_in,
                              void* d_out_v, int out_size, void* d_ws, size_t ws_size,
                              hipStream_t stream) {
    const float* kn    = (const float*)d_in[0];
    const float* exer  = (const float*)d_in[1];
    const float* stu   = (const float*)d_in[2];
    const float* W     = (const float*)d_in[3];
    const float* a_src = (const float*)d_in[4];
    const float* a_dst = (const float*)d_in[5];
    const float* b_gat = (const float*)d_in[6];
    const float* attnw = (const float*)d_in[7];
    const float* attnb = (const float*)d_in[8];
    const int*   srcKE = (const int*)d_in[9];    // edges_ke[0]: 2048 entries
    const int*   srcUE = (const int*)d_in[11];   // edges_ue[0]: 150000 entries
    float* out = (float*)d_out_v;

    // workspace layout (floats first, then bf16 blocks)
    float* waS   = (float*)d_ws;          // 2048
    float* waD   = waS + 2048;            // 2048
    float* asE   = waD + 2048;            // 24
    float* adE   = asE + 24;              // 24
    float* as_k1 = adE + 24;              // 512
    float* emaxC = as_k1 + 512;           // 6
    float* denC  = emaxC + 6;             // 6
    float* numC  = denC + 6;              // 3072
    float* asStu = numC + 3072;           // 50000
    float* Bmat  = asStu + 50000;         // 3072
    float* Cmat  = Bmat + 3072;           // 3072
    u16*   zB    = (u16*)(Cmat + 3072);   // 3072
    u16*   zC    = zB + 3072;             // 3072
    u16*   Wt    = zC + 3072;             // 4*262144

    float* out_kn   = out;                 // 512*512
    float* out_exer = out + 262144;        // 6*512
    float* out_stu  = out + 265216;        // 50000*512

    k_cvt  <<<4096, 256, 0, stream>>>(W, Wt);
    k_wa   <<<2048, 256, 0, stream>>>(W, a_src, a_dst, waS, waD);
    k_small<<<536, 256, 0, stream>>>(kn, exer, waS, waD, as_k1, asE, adE);

    // GAT2 -> zB ; GAT4 -> zC
    k_zB   <<<6, 256, 0, stream>>>(kn, exer, srcKE, as_k1, asE + 6, adE + 6, zB);
    k_initc<<<12, 256, 0, stream>>>(emaxC, denC, numC);
    k_asstu<<<6250, 256, 0, stream>>>(stu, waS + 3 * 512, asStu);
    k_cmax <<<96, 256, 0, stream>>>(srcUE, asStu, asE + 18, adE + 18, emaxC);
    k_cacc <<<384, 256, 0, stream>>>(srcUE, stu, asStu, adE + 18, emaxC, denC, numC);
    k_cfix <<<6, 256, 0, stream>>>(exer, asE + 18, adE + 18, emaxC, denC, numC, zC);

    // GAT1 fused GEMM: kn_out = kn + z_k @ W0 + b0   (softmax over 4 edges + self)
    gemm_tile<<<dim3(4, 4), 256, 0, stream>>>(kn, nullptr, Wt, b_gat, kn, exer,
                                              srcKE, asE, waS, waD, out_kn, 512, 4, 1);
    // GAT3 fused GEMM: stu_out = stu + z_u @ W2 + b2
    gemm_tile<<<dim3(391, 4), 256, 0, stream>>>(stu, nullptr, Wt + 2 * 262144, b_gat + 1024, stu, exer,
                                                srcUE, asE + 12, waS + 1024, waD + 1024,
                                                out_stu, 50000, 3, 1);
    // B = zB @ W1 + b1 ; C = zC @ W3 + b3
    gemm_tile<<<dim3(1, 4), 256, 0, stream>>>(exer, zB, Wt + 262144, b_gat + 512, nullptr, exer,
                                              nullptr, nullptr, nullptr, nullptr, Bmat, 6, 0, 0);
    gemm_tile<<<dim3(1, 4), 256, 0, stream>>>(exer, zC, Wt + 3 * 262144, b_gat + 1536, nullptr, exer,
                                              nullptr, nullptr, nullptr, nullptr, Cmat, 6, 0, 0);

    k_exer<<<6, 256, 0, stream>>>(exer, Bmat, Cmat, attnw, attnb, out_exer);
}
